// BahdanauAttention_26938034880922
// MI455X (gfx1250) — matrix operand order repaired and checked by reading, not compiled, not run on hardware
//
#include <hip/hip_runtime.h>
#include <hip/hip_bf16.h>
#include <math.h>

// Problem sizes (fixed by the reference)
#define Bq 32
#define Sq 2048
#define Hq 1024
#define Uq 1024

#define LDA 1032            // padded LDS row length (bf16 elems): 2064B stride,
                            // 516 dwords % 64 banks = 4 -> conflict-free b128 reads
#define A_ELEMS (64 * LDA)  // 64 s-rows staged per workgroup
#define A_BYTES (A_ELEMS * 2)
#define SMEM_BYTES (A_BYTES + 64 * 4)  // + ssum[64]

typedef __bf16 bf16;
typedef __attribute__((ext_vector_type(16))) __bf16 v16bf;
typedef __attribute__((ext_vector_type(8)))  __bf16 v8bf;
typedef __attribute__((ext_vector_type(2)))  __bf16 v2bf;
typedef __attribute__((ext_vector_type(8)))  float  v8f;

// ---------------------------------------------------------------------------
// Kernel 0: W1 [H,U] f32 row-major  ->  W1T [U,H] bf16 row-major (transpose+cvt)
// ---------------------------------------------------------------------------
__global__ __launch_bounds__(256) void w1_to_bf16T(const float* __restrict__ W1,
                                                   bf16* __restrict__ W1T) {
  __shared__ float tile[32][33];
  const int u0 = blockIdx.x * 32;
  const int h0 = blockIdx.y * 32;
  const int tx = threadIdx.x;   // 0..31
  const int ty = threadIdx.y;   // 0..7
  for (int j = 0; j < 4; ++j) {
    int r = ty + j * 8;
    tile[r][tx] = W1[(size_t)(h0 + r) * Uq + (u0 + tx)];  // coalesced in tx
  }
  __syncthreads();
  for (int j = 0; j < 4; ++j) {
    int r = ty + j * 8;
    W1T[(size_t)(u0 + r) * Hq + (h0 + tx)] = (bf16)tile[tx][r];  // coalesced in tx
  }
}

// ---------------------------------------------------------------------------
// Kernel 1: proj_q[b,u] = sum_h query[b,h] * W2[h,u] + b2[u]   (tiny)
// ---------------------------------------------------------------------------
__global__ __launch_bounds__(256) void projq_kernel(const float* __restrict__ query,
                                                    const float* __restrict__ W2,
                                                    const float* __restrict__ b2,
                                                    float* __restrict__ projq) {
  const int b = blockIdx.y;
  const int u = blockIdx.x * 256 + threadIdx.x;
  float acc = b2[u];
  const float* q = query + (size_t)b * Hq;
  for (int h = 0; h < Hq; ++h) {
    acc += q[h] * W2[(size_t)h * Uq + u];  // W2 coalesced in u; q[h] scalar
  }
  projq[(size_t)b * Uq + u] = acc;
}

// ---------------------------------------------------------------------------
// Kernel 2 (WMMA workhorse):
//   score[b,s] = sum_u tanh( (values[b,s,:] @ W1)[u] + b1[u] + projq[b,u] ) * V[u] + bv
//
// Workgroup: 1024 threads = 32 waves; tile = 64 s-rows x 1024 u-cols for one b.
// Stage the full 64x1024 A panel as bf16 in LDS once (132KB of the 320KB WGP
// LDS), then run the whole K-loop with zero barriers.
// Wave w: s-group = w>>4 (32 rows as 2 A-frags), u-group = w&15 (64 cols as
// 4 WMMA tiles). Each B fragment is loaded once and reused by 2 WMMAs:
// per k-step per wave = 4 ds_load_b128 + 8 global_load_b128 + 8 wmma.
// ---------------------------------------------------------------------------
__global__ __launch_bounds__(1024) void score_kernel(const float* __restrict__ values,
                                                     const bf16*  __restrict__ W1T,
                                                     const float* __restrict__ b1,
                                                     const float* __restrict__ projq,
                                                     const float* __restrict__ Vvec,
                                                     const float* __restrict__ bv,
                                                     float* __restrict__ score) {
  extern __shared__ char smem[];
  bf16*  As   = (bf16*)smem;              // [64][LDA]
  float* ssum = (float*)(smem + A_BYTES); // [64]

  const int b    = blockIdx.y;
  const int s0   = blockIdx.x * 64;
  const int tid  = threadIdx.x;
  const int wave = tid >> 5;      // 0..31
  const int lane = tid & 31;
  const int sgrp = wave >> 4;     // which 32-row s-group (0/1)
  const int ugrp = wave & 15;     // which 64-col u-group (0..15)
  const int l15  = lane & 15;
  const int hi   = lane >> 4;     // lane half (0/1)

  const float* vrow_base = values + ((size_t)b * Sq + s0) * Hq;

  // --- one-time cooperative stage of A panel: 64 rows x 1024 K, f32 -> bf16 ---
  for (int i = 0; i < 32; ++i) {
    const int e2 = (tid + 1024 * i) * 2;       // even element index 0..65534
    const int r = e2 >> 10;                    // 0..63
    const int k = e2 & 1023;                   // even
    const float2 v = *(const float2*)(vrow_base + (size_t)r * Hq + k);
    v2bf p;
    p.x = (bf16)v.x;
    p.y = (bf16)v.y;
    *(v2bf*)(As + (size_t)r * LDA + k) = p;    // 4B-aligned ds_store_b32
  }
  __syncthreads();

  const v8f vzero = {0.f, 0.f, 0.f, 0.f, 0.f, 0.f, 0.f, 0.f};
  v8f acc[2][4];
  for (int sf = 0; sf < 2; ++sf)
    for (int t = 0; t < 4; ++t) acc[sf][t] = vzero;

  // B base: column (ugrp*64 + l15) of W1 == row of W1T; K offset hi*8.
  const bf16* bbase = W1T + (size_t)(ugrp * 64 + l15) * Hq + (hi ? 8 : 0);
  const int ac = hi ? 8 : 0;           // A-frag K offset for this lane half
  const int arow0 = sgrp * 32 + l15;   // first A-frag row for this lane

  // --- K loop: no barriers, fully pipelineable ---
  for (int k0 = 0; k0 < Hq; k0 += 32) {
    // A fragments (16-bit A 16x32 layout):
    // lanes 0-15: K 0-7 then 16-23; lanes 16-31: K 8-15 then 24-31
    v16bf afrag[2];
    for (int sf = 0; sf < 2; ++sf) {
      const bf16* ap = As + (size_t)(arow0 + sf * 16) * LDA + k0 + ac;
      v8bf alo = *(const v8bf*)ap;
      v8bf ahi = *(const v8bf*)(ap + 16);
      afrag[sf] = __builtin_shufflevector(alo, ahi,
          0, 1, 2, 3, 4, 5, 6, 7, 8, 9, 10, 11, 12, 13, 14, 15);
    }
    // 4 B fragments, each consumed by 2 WMMAs (both s-groups of this wave)
    for (int t = 0; t < 4; ++t) {
      const bf16* bp = bbase + (size_t)t * 16 * Hq + k0;
      v8bf blo = *(const v8bf*)bp;         // K +0..7  (of this lane-half)
      v8bf bhi = *(const v8bf*)(bp + 16);  // K +16..23
      __builtin_prefetch((const void*)(bp + 32), 0, 1);  // next k-step's chunk
      v16bf bfrag = __builtin_shufflevector(blo, bhi,
          0, 1, 2, 3, 4, 5, 6, 7, 8, 9, 10, 11, 12, 13, 14, 15);
      acc[0][t] = __builtin_amdgcn_wmma_f32_16x16x32_bf16(
          false, afrag[0], false, bfrag, (short)0, acc[0][t], false, false);
      acc[1][t] = __builtin_amdgcn_wmma_f32_16x16x32_bf16(
          false, afrag[1], false, bfrag, (short)0, acc[1][t], false, false);
    }
  }

  // --- epilogue: tanh(acc + b1 + projq) dot V, reduce over u per s-row ---
  if (tid < 64) ssum[tid] = 0.f;
  __syncthreads();

  float psum[2][8];
  for (int sf = 0; sf < 2; ++sf)
    for (int i = 0; i < 8; ++i) psum[sf][i] = 0.f;

  for (int t = 0; t < 4; ++t) {
    const int u = ugrp * 64 + t * 16 + l15;   // column of this lane
    const float addv = b1[u] + projq[(size_t)b * Uq + u];
    const float vc = Vvec[u];
    for (int sf = 0; sf < 2; ++sf) {
      for (int i = 0; i < 8; ++i) {
        float x = tanhf(acc[sf][t][i] + addv);
        psum[sf][i] += x * vc;
      }
    }
  }
  // C/D layout: VGPR i -> row M = i + 8*hi within a 16-row fragment
  for (int sf = 0; sf < 2; ++sf) {
    const int rbase = sgrp * 32 + sf * 16 + hi * 8;
    for (int i = 0; i < 8; ++i) atomicAdd(&ssum[rbase + i], psum[sf][i]);
  }
  __syncthreads();

  if (tid < 64) score[(size_t)b * Sq + s0 + tid] = ssum[tid] + bv[0];
}

// ---------------------------------------------------------------------------
// Kernel 3: softmax over the S axis for each b; writes attention weights
// ---------------------------------------------------------------------------
__global__ __launch_bounds__(256) void softmax_kernel(const float* __restrict__ score,
                                                      float* __restrict__ attn) {
  const int b = blockIdx.x;
  const int tid = threadIdx.x;
  __shared__ float red[256];
  const float* s = score + (size_t)b * Sq;

  float m = -1e30f;
  for (int i = tid; i < Sq; i += 256) m = fmaxf(m, s[i]);
  red[tid] = m;
  __syncthreads();
  for (int off = 128; off > 0; off >>= 1) {
    if (tid < off) red[tid] = fmaxf(red[tid], red[tid + off]);
    __syncthreads();
  }
  m = red[0];
  __syncthreads();

  float sum = 0.f;
  for (int i = tid; i < Sq; i += 256) sum += expf(s[i] - m);
  red[tid] = sum;
  __syncthreads();
  for (int off = 128; off > 0; off >>= 1) {
    if (tid < off) red[tid] += red[tid + off];
    __syncthreads();
  }
  const float inv = 1.f / red[0];

  for (int i = tid; i < Sq; i += 256) {
    attn[(size_t)b * Sq + i] = expf(s[i] - m) * inv;
  }
}

// ---------------------------------------------------------------------------
// Kernel 4: context[b,h] = sum_s attn[b,s] * values[b,s,h]   (memory-bound)
// ---------------------------------------------------------------------------
__global__ __launch_bounds__(256) void context_kernel(const float* __restrict__ values,
                                                      const float* __restrict__ attn,
                                                      float* __restrict__ ctx) {
  const int b = blockIdx.y;
  const int h = blockIdx.x * 256 + threadIdx.x;
  __shared__ float wbuf[256];
  float acc = 0.f;
  const float* vb = values + (size_t)b * Sq * Hq;
  for (int sblk = 0; sblk < Sq; sblk += 256) {
    __syncthreads();
    wbuf[threadIdx.x] = attn[(size_t)b * Sq + sblk + threadIdx.x];
    __syncthreads();
    for (int j = 0; j < 256; ++j) {
      acc += wbuf[j] * vb[(size_t)(sblk + j) * Hq + h];  // coalesced in h
    }
  }
  ctx[(size_t)b * Hq + h] = acc;
}

// ---------------------------------------------------------------------------
// Host launcher
// ---------------------------------------------------------------------------
extern "C" void kernel_launch(void* const* d_in, const int* in_sizes, int n_in,
                              void* d_out, int out_size, void* d_ws, size_t ws_size,
                              hipStream_t stream) {
  const float* query  = (const float*)d_in[0];  // [B,H]
  const float* values = (const float*)d_in[1];  // [B,S,H]
  const float* W1     = (const float*)d_in[2];  // [H,U]
  const float* b1     = (const float*)d_in[3];  // [U]
  const float* W2     = (const float*)d_in[4];  // [H,U]
  const float* b2     = (const float*)d_in[5];  // [U]
  const float* Vv     = (const float*)d_in[6];  // [U,1]
  const float* bv     = (const float*)d_in[7];  // [1]

  float* out  = (float*)d_out;
  float* ctx  = out;                       // [B,H]   (first tuple output)
  float* attn = out + (size_t)Bq * Hq;     // [B,S,1] (second tuple output)

  char* ws = (char*)d_ws;
  bf16*  W1T   = (bf16*)ws;                                        // 2 MB
  float* projq = (float*)(ws + (size_t)Uq * Hq * sizeof(bf16));    // 128 KB
  float* score = (float*)(ws + (size_t)Uq * Hq * sizeof(bf16)
                             + (size_t)Bq * Uq * sizeof(float));   // 256 KB

  w1_to_bf16T<<<dim3(Uq / 32, Hq / 32), dim3(32, 8), 0, stream>>>(W1, W1T);
  projq_kernel<<<dim3(Uq / 256, Bq), 256, 0, stream>>>(query, W2, b2, projq);
  // 64 s-rows per workgroup, dynamic LDS = 132.3 KB (of 320 KB per WGP)
  score_kernel<<<dim3(Sq / 64, Bq), 1024, SMEM_BYTES, stream>>>(
      values, W1T, b1, projq, Vv, bv, score);
  softmax_kernel<<<Bq, 256, 0, stream>>>(score, attn);
  context_kernel<<<dim3(Hq / 256, Bq), 256, 0, stream>>>(values, attn, ctx);
}
